// PedTRDeformTransformer_3882650436954
// MI455X (gfx1250) — matrix operand order repaired and checked by reading, not compiled, not run on hardware
//
#include <hip/hip_runtime.h>
#include <hip/hip_bf16.h>
#include <math.h>

// ---------------------------------------------------------------------------
// Problem constants
// ---------------------------------------------------------------------------
#define E_DIM   512
#define NH      8
#define HD      64
#define P_PTS   4
#define Q_NUM   900
#define CAMS    7
#define DEPTH   4
#define QP      (Q_NUM * P_PTS)       // 3600
#define TOK     (QP * CAMS)           // 25200
#define FH      68
#define FW      120
#define IMG_H   1080.0f
#define IMG_W   1920.0f
#define GRID0   480.0f
#define GRID1   1440.0f
#define NEGINF  (-1e9f)

typedef __attribute__((ext_vector_type(16))) _Float16 v16h;
typedef __attribute__((ext_vector_type(8)))  float    v8f;

__device__ __forceinline__ unsigned int pkh2(float a, float b) {
    union { _Float16 h[2]; unsigned int u; } t;
    t.h[0] = (_Float16)a; t.h[1] = (_Float16)b;
    return t.u;
}

// ---------------------------------------------------------------------------
// WMMA GEMM:  C[M,N] = scale * (A[M,K] x B) + bias (+ relu) (+ resid)
//   TRANSB=1 : B is [N,K] row-major (weights; C = A * B^T)
//   TRANSB=0 : B is [K,N] row-major (C = A * B)
// BM=128, BN=64, BK=32; 256 threads = 8 waves; each wave -> 32x32 of C
// (4x v_wmma_f32_16x16x32_f16 per k-step, f32 accumulate).
// LDS layout: As[m][k], Bs[n][k]  (both k-contiguous -> dword fragment loads)
// ---------------------------------------------------------------------------
#define BM 128
#define BN 64
#define BK 32
#define KP (BK + 8)    // padded row length (halfs); 80B rows keep 16B alignment

template<int TRANSB, int RELU, int RESID>
__global__ void __launch_bounds__(256)
gemm_wmma(const float* __restrict__ A, const float* __restrict__ B,
          const float* __restrict__ bias, const float* __restrict__ resid,
          float* __restrict__ C,
          int M, int N, int K, int lda, int ldb, int ldc, float scale)
{
    __shared__ __align__(16) _Float16 As[BM][KP];
    __shared__ __align__(16) _Float16 Bs[BN][KP];

    const int tid  = threadIdx.x;
    const int bm   = blockIdx.y * BM;
    const int bn   = blockIdx.x * BN;
    const int wave = tid >> 5;
    const int lane = tid & 31;
    const int wm   = wave >> 1;      // 0..3 : 32-row slice of M
    const int wn   = wave & 1;       // 0..1 : 32-col slice of N

    v8f acc00 = {}, acc01 = {}, acc10 = {}, acc11 = {};

    // staging coordinates
    const int arow = tid >> 1;             // 0..127
    const int acol = (tid & 1) * 16;       // 0 or 16
    const int brow = tid >> 2;             // 0..63   (TRANSB staging: n index)
    const int bcol = (tid & 3) * 8;        // k chunk
    const int nrow = tid >> 3;             // 0..31   (!TRANSB staging: k index)
    const int ncol = (tid & 7) * 8;        // n chunk

    for (int kt = 0; kt < K; kt += BK) {
        const bool fullK = (kt + BK) <= K;
        // ---------------- stage A tile (128 x 32) ----------------
        {
            int gr = bm + arow;
            if (fullK && (bm + BM) <= M) {
                const float4* p = (const float4*)(A + (size_t)gr * lda + kt + acol);
                float4 f0 = p[0], f1 = p[1], f2 = p[2], f3 = p[3];
                if (kt + BK < K)
                    __builtin_prefetch(A + (size_t)gr * lda + kt + BK + acol, 0, 1);
                uint4 s0, s1;
                s0.x = pkh2(f0.x, f0.y); s0.y = pkh2(f0.z, f0.w);
                s0.z = pkh2(f1.x, f1.y); s0.w = pkh2(f1.z, f1.w);
                s1.x = pkh2(f2.x, f2.y); s1.y = pkh2(f2.z, f2.w);
                s1.z = pkh2(f3.x, f3.y); s1.w = pkh2(f3.z, f3.w);
                *(uint4*)&As[arow][acol]     = s0;
                *(uint4*)&As[arow][acol + 8] = s1;
            } else {
#pragma unroll
                for (int u = 0; u < 16; ++u) {
                    int gc  = kt + acol + u;
                    float v = (gr < M && gc < K) ? A[(size_t)gr * lda + gc] : 0.0f;
                    As[arow][acol + u] = (_Float16)v;
                }
            }
        }
        // ---------------- stage B tile into Bs[n][k] ----------------
        if (TRANSB) {
            int gn = bn + brow;
            if (fullK && (bn + BN) <= N) {
                const float4* p = (const float4*)(B + (size_t)gn * ldb + kt + bcol);
                float4 f0 = p[0], f1 = p[1];
                if (kt + BK < K)
                    __builtin_prefetch(B + (size_t)gn * ldb + kt + BK + bcol, 0, 1);
                uint4 s0;
                s0.x = pkh2(f0.x, f0.y); s0.y = pkh2(f0.z, f0.w);
                s0.z = pkh2(f1.x, f1.y); s0.w = pkh2(f1.z, f1.w);
                *(uint4*)&Bs[brow][bcol] = s0;
            } else {
#pragma unroll
                for (int u = 0; u < 8; ++u) {
                    int gk  = kt + bcol + u;
                    float v = (gn < N && gk < K) ? B[(size_t)gn * ldb + gk] : 0.0f;
                    Bs[brow][bcol + u] = (_Float16)v;
                }
            }
        } else {
            int gk = kt + nrow;
            if (gk < K && (bn + BN) <= N) {
                const float4* p = (const float4*)(B + (size_t)gk * ldb + bn + ncol);
                float4 f0 = p[0], f1 = p[1];
                Bs[ncol + 0][nrow] = (_Float16)f0.x;
                Bs[ncol + 1][nrow] = (_Float16)f0.y;
                Bs[ncol + 2][nrow] = (_Float16)f0.z;
                Bs[ncol + 3][nrow] = (_Float16)f0.w;
                Bs[ncol + 4][nrow] = (_Float16)f1.x;
                Bs[ncol + 5][nrow] = (_Float16)f1.y;
                Bs[ncol + 6][nrow] = (_Float16)f1.z;
                Bs[ncol + 7][nrow] = (_Float16)f1.w;
            } else {
#pragma unroll
                for (int u = 0; u < 8; ++u) {
                    int gn  = bn + ncol + u;
                    float v = (gk < K && gn < N) ? B[(size_t)gk * ldb + gn] : 0.0f;
                    Bs[ncol + u][nrow] = (_Float16)v;
                }
            }
        }
        __syncthreads();

        // ---------------- fragments (dword reads from LDS) ----------------
        // A 16x32 f16: lane m=l&15; K {0-7,16-23} (lanes 0-15) / {8-15,24-31}
        //   dword idx: j=0..3 -> 4*half + j ; j=4..7 -> 8 + 4*half + (j-4)
        // B 32x16 f16: lane n=l&15; K {0-7,16-23} (lanes 0-15) / {8-15,24-31}
        //   dword idx: j=0..3 -> 4*half + j ; j=4..7 -> 8 + 4*half + (j-4)
        const int half = lane >> 4;
        const int l15  = lane & 15;
        const unsigned int* ar0 = (const unsigned int*)&As[wm * 32 + l15][0];
        const unsigned int* ar1 = (const unsigned int*)&As[wm * 32 + 16 + l15][0];
        const unsigned int* br0 = (const unsigned int*)&Bs[wn * 32 + l15][0];
        const unsigned int* br1 = (const unsigned int*)&Bs[wn * 32 + 16 + l15][0];
        union { v16h v; unsigned int u[8]; } a0, a1, b0, b1;
        const int ah = half * 4;
        const int bh = half * 8;
#pragma unroll
        for (int j = 0; j < 4; ++j) {
            a0.u[j]     = ar0[ah + j];
            a0.u[4 + j] = ar0[8 + ah + j];
            a1.u[j]     = ar1[ah + j];
            a1.u[4 + j] = ar1[8 + ah + j];
        }
#pragma unroll
        for (int j = 0; j < 8; ++j) {
            b0.u[j] = br0[(j >> 2) * 8 + ah + (j & 3)];
            b1.u[j] = br1[(j >> 2) * 8 + ah + (j & 3)];
        }
        acc00 = __builtin_amdgcn_wmma_f32_16x16x32_f16(false, a0.v, false, b0.v,
                                                       (short)0, acc00, false, false);
        acc01 = __builtin_amdgcn_wmma_f32_16x16x32_f16(false, a0.v, false, b1.v,
                                                       (short)0, acc01, false, false);
        acc10 = __builtin_amdgcn_wmma_f32_16x16x32_f16(false, a1.v, false, b0.v,
                                                       (short)0, acc10, false, false);
        acc11 = __builtin_amdgcn_wmma_f32_16x16x32_f16(false, a1.v, false, b1.v,
                                                       (short)0, acc11, false, false);
        __syncthreads();
    }

    // ---- epilogue: C layout: VGPR r -> M = r + 8*(lane>=16), N = lane&15 ----
    const int l15   = lane & 15;
    const int nc0   = bn + wn * 32 + l15;
    const int nc1   = nc0 + 16;
    const int mb0   = bm + wm * 32 + (lane >> 4) * 8;
    const int mb1   = mb0 + 16;
    const v8f* accs[4] = { &acc00, &acc01, &acc10, &acc11 };
    const int  mbs[4]  = { mb0, mb0, mb1, mb1 };
    const int  ncs[4]  = { nc0, nc1, nc0, nc1 };
#pragma unroll
    for (int s = 0; s < 4; ++s) {
        const v8f a = *accs[s];
        const int nc = ncs[s];
        if (nc >= N) continue;
#pragma unroll
        for (int r = 0; r < 8; ++r) {
            int m = mbs[s] + r;
            if (m >= M) continue;
            float v = a[r] * scale;
            if (bias)  v += bias[nc];
            if (RELU)  v = fmaxf(v, 0.0f);
            if (RESID) v += resid[(size_t)m * ldc + nc];
            C[(size_t)m * ldc + nc] = v;
        }
    }
}

// ---------------------------------------------------------------------------
// Elementwise / reduction helpers
// ---------------------------------------------------------------------------
__global__ void k_add2(const float* a, const float* b, float* y, int n) {
    int i = blockIdx.x * blockDim.x + threadIdx.x;
    if (i < n) y[i] = a[i] + b[i];
}

// row LayerNorm with optional two pre-adds and optional post-ReLU
__global__ void k_layernorm(const float* __restrict__ x, const float* __restrict__ a1,
                            const float* __restrict__ a2, const float* __restrict__ g,
                            const float* __restrict__ b, float* __restrict__ y,
                            int E, int relu)
{
    const int row = blockIdx.x;
    const int t   = threadIdx.x;
    const size_t base = (size_t)row * E;
    __shared__ float rs[256];
    __shared__ float rq[256];
    float s = 0.0f, s2 = 0.0f;
    for (int i = t; i < E; i += 256) {
        float v = x[base + i];
        if (a1) v += a1[base + i];
        if (a2) v += a2[base + i];
        y[base + i] = v;
        s += v; s2 += v * v;
    }
    rs[t] = s; rq[t] = s2;
    __syncthreads();
    for (int st = 128; st > 0; st >>= 1) {
        if (t < st) { rs[t] += rs[t + st]; rq[t] += rq[t + st]; }
        __syncthreads();
    }
    float mean = rs[0] / (float)E;
    float var  = rq[0] / (float)E - mean * mean;
    float inv  = rsqrtf(var + 1e-5f);
    for (int i = t; i < E; i += 256) {
        float v = (y[base + i] - mean) * inv * g[i] + b[i];
        if (relu) v = fmaxf(v, 0.0f);
        y[base + i] = v;
    }
}

__global__ void k_softmax_rows(float* __restrict__ x, int cols) {
    const int row = blockIdx.x;
    const int t   = threadIdx.x;
    float* xr = x + (size_t)row * cols;
    __shared__ float red[256];
    float mx = -3.4e38f;
    for (int i = t; i < cols; i += 256) mx = fmaxf(mx, xr[i]);
    red[t] = mx; __syncthreads();
    for (int st = 128; st > 0; st >>= 1) {
        if (t < st) red[t] = fmaxf(red[t], red[t + st]);
        __syncthreads();
    }
    mx = red[0]; __syncthreads();
    float sum = 0.0f;
    for (int i = t; i < cols; i += 256) {
        float e = expf(xr[i] - mx);
        xr[i] = e; sum += e;
    }
    red[t] = sum; __syncthreads();
    for (int st = 128; st > 0; st >>= 1) {
        if (t < st) red[t] += red[t + st];
        __syncthreads();
    }
    float inv = 1.0f / red[0];
    for (int i = t; i < cols; i += 256) xr[i] *= inv;
}

// tiny fully-connected (used for N=8 / N=4 heads of the MLPs)
__global__ void k_small_fc(const float* __restrict__ x, const float* __restrict__ w,
                           const float* __restrict__ b, float* __restrict__ y,
                           int M, int N, int K)
{
    int idx = blockIdx.x * blockDim.x + threadIdx.x;
    if (idx >= M * N) return;
    int m = idx / N, n = idx % N;
    float s = b ? b[n] : 0.0f;
    const float* xr = x + (size_t)m * K;
    const float* wr = w + (size_t)n * K;
    for (int k = 0; k < K; ++k) s += xr[k] * wr[k];
    y[idx] = s;
}

__device__ __forceinline__ float inv_sig(float x) {
    x = fminf(fmaxf(x, 0.0f), 1.0f);
    float a = fmaxf(x, 1e-5f);
    float b = fmaxf(1.0f - x, 1e-5f);
    return logf(a / b);
}

// projection: rp_d -> camera-plane coords + validity
__global__ void k_project(const float* __restrict__ ref, const float* __restrict__ off,
                          const float* __restrict__ projm,
                          float* __restrict__ camxy, unsigned char* __restrict__ validb)
{
    int j = blockIdx.x * blockDim.x + threadIdx.x;
    if (j >= QP) return;
    // reference quirk: tile(ref,(P,1)) -> ref[j % Q]; offset from reshape -> off[j]
    float rx = ref[(j % Q_NUM) * 2 + 0];
    float ry = ref[(j % Q_NUM) * 2 + 1];
    float px = 1.0f / (1.0f + expf(-(inv_sig(rx) + off[j * 2 + 0]))) * GRID0;
    float py = 1.0f / (1.0f + expf(-(inv_sig(ry) + off[j * 2 + 1]))) * GRID1;
    for (int c = 0; c < CAMS; ++c) {
        const float* M = projm + c * 9;
        float ic0 = M[0] * px + M[1] * py + M[2];
        float ic1 = M[3] * px + M[4] * py + M[5];
        float ic2 = M[6] * px + M[7] * py + M[8];
        float z   = fmaxf(ic2, 1e-5f);
        float xx  = ic0 / z, yy = ic1 / z;
        float cx  = (xx / IMG_W - 0.5f) * 2.0f;
        float cy  = (yy / IMG_H - 0.5f) * 2.0f;
        int v = (ic2 > 1e-5f) && (cx > -1.0f) && (cx < 1.0f) && (cy > -1.0f) && (cy < 1.0f);
        camxy[((size_t)c * QP + j) * 2 + 0] = cx;
        camxy[((size_t)c * QP + j) * 2 + 1] = cy;
        validb[(size_t)c * QP + j] = (unsigned char)v;
    }
}

// bilinear grid sample (align_corners=False, zero padding) + nan_to_num
__global__ void k_sample(const float* __restrict__ feat, const float* __restrict__ camxy,
                         float* __restrict__ x)
{
    int b = blockIdx.x;           // c*QP + j
    int c = b / QP, j = b % QP;
    float gx = camxy[(size_t)b * 2 + 0];
    float gy = camxy[(size_t)b * 2 + 1];
    float fx = (gx + 1.0f) * (FW * 0.5f) - 0.5f;
    float fy = (gy + 1.0f) * (FH * 0.5f) - 0.5f;
    float fx0 = floorf(fx), fy0 = floorf(fy);
    int x0 = (int)fx0, y0 = (int)fy0;
    int x1 = x0 + 1,  y1 = y0 + 1;
    float wx1 = fx - fx0, wy1 = fy - fy0;
    float wx0 = 1.0f - wx1, wy0 = 1.0f - wy1;
    float w00 = (y0 >= 0 && y0 < FH && x0 >= 0 && x0 < FW) ? wy0 * wx0 : 0.0f;
    float w01 = (y0 >= 0 && y0 < FH && x1 >= 0 && x1 < FW) ? wy0 * wx1 : 0.0f;
    float w10 = (y1 >= 0 && y1 < FH && x0 >= 0 && x0 < FW) ? wy1 * wx0 : 0.0f;
    float w11 = (y1 >= 0 && y1 < FH && x1 >= 0 && x1 < FW) ? wy1 * wx1 : 0.0f;
    int cx0 = min(max(x0, 0), FW - 1), cx1 = min(max(x1, 0), FW - 1);
    int cy0 = min(max(y0, 0), FH - 1), cy1 = min(max(y1, 0), FH - 1);
    const float* fb = feat + (size_t)c * E_DIM * FH * FW;
    for (int e = threadIdx.x; e < E_DIM; e += blockDim.x) {
        const float* fe = fb + (size_t)e * FH * FW;
        float v = fe[cy0 * FW + cx0] * w00 + fe[cy0 * FW + cx1] * w01
                + fe[cy1 * FW + cx0] * w10 + fe[cy1 * FW + cx1] * w11;
        if (v != v) v = 0.0f;                         // nan_to_num
        v = fminf(fmaxf(v, -3.4e38f), 3.4e38f);
        x[((size_t)j * CAMS + c) * E_DIM + e] = v;
    }
}

// per-(sequence, head) 7x7 masked attention for the ImgFeatureTransformer
__global__ void k_tattn(const float* __restrict__ qkv, const unsigned char* __restrict__ validb,
                        float* __restrict__ out)
{
    int b = blockIdx.x;           // j*NH + h
    int j = b >> 3;
    int h = b & 7;
    int t = threadIdx.x;          // 0..63
    __shared__ float qs[CAMS][HD];
    __shared__ float ks[CAMS][HD];
    __shared__ float vs[CAMS][HD];
    __shared__ float sc[CAMS][8];
    __shared__ float pr[CAMS][8];
    for (int l = 0; l < CAMS; ++l) {
        size_t base = ((size_t)(j * CAMS + l)) * (3 * E_DIM) + (size_t)h * HD + t;
        qs[l][t] = qkv[base];
        ks[l][t] = qkv[base + E_DIM];
        vs[l][t] = qkv[base + 2 * E_DIM];
    }
    __syncthreads();
    // reference mask batch permutation: q' = ((j%Q)*NH + h) % Q ; p = j/Q
    int r  = j % Q_NUM;
    int p  = j / Q_NUM;
    int mq = ((r * NH + h) % Q_NUM) * P_PTS + p;
    if (t < CAMS * CAMS) {
        int l = t / CAMS, s = t % CAMS;
        float acc = 0.0f;
        for (int d = 0; d < HD; ++d) acc += qs[l][d] * ks[s][d];
        acc *= 0.125f;                         // 1/sqrt(64)
        if (!validb[(size_t)s * QP + mq]) acc = NEGINF;
        sc[l][s] = acc;
    }
    __syncthreads();
    if (t < CAMS) {
        float mx = -3.4e38f;
        for (int s = 0; s < CAMS; ++s) mx = fmaxf(mx, sc[t][s]);
        float den = 0.0f;
        for (int s = 0; s < CAMS; ++s) { float e = expf(sc[t][s] - mx); pr[t][s] = e; den += e; }
        float inv = 1.0f / den;
        for (int s = 0; s < CAMS; ++s) pr[t][s] *= inv;
    }
    __syncthreads();
    for (int l = 0; l < CAMS; ++l) {
        float o = 0.0f;
        for (int s = 0; s < CAMS; ++s) o += pr[l][s] * vs[s][t];
        out[((size_t)(j * CAMS + l)) * E_DIM + (size_t)h * HD + t] = o;
    }
}

// out = mean_p( attn_w[q,p] * mean_c x[(q*4+p)*7+c, :] )
__global__ void k_reduce_out(const float* __restrict__ x, const float* __restrict__ aw,
                             float* __restrict__ outq)
{
    int idx = blockIdx.x * blockDim.x + threadIdx.x;   // q*E + e
    if (idx >= Q_NUM * E_DIM) return;
    int q = idx >> 9, e = idx & (E_DIM - 1);
    float acc = 0.0f;
    for (int p = 0; p < P_PTS; ++p) {
        int j = q * P_PTS + p;
        float s = 0.0f;
        for (int c = 0; c < CAMS; ++c) s += x[((size_t)j * CAMS + c) * E_DIM + e];
        acc += (s * (1.0f / CAMS)) * aw[q * P_PTS + p];
    }
    outq[idx] = acc * 0.25f;
}

// pe first layer: K=2
__global__ void k_pe1(const float* __restrict__ ref, const float* __restrict__ w,
                      const float* __restrict__ b, float* __restrict__ y)
{
    int idx = blockIdx.x * blockDim.x + threadIdx.x;
    if (idx >= Q_NUM * E_DIM) return;
    int q = idx >> 9, e = idx & (E_DIM - 1);
    float ix = inv_sig(ref[q * 2 + 0]);
    float iy = inv_sig(ref[q * 2 + 1]);
    y[idx] = ix * w[e * 2 + 0] + iy * w[e * 2 + 1] + b[e];
}

// ---------------------------------------------------------------------------
// Orchestration
// ---------------------------------------------------------------------------
extern "C" void kernel_launch(void* const* d_in, const int* in_sizes, int n_in,
                              void* d_out, int out_size, void* d_ws, size_t ws_size,
                              hipStream_t stream)
{
    (void)in_sizes; (void)n_in; (void)out_size; (void)ws_size;
    const float* img_feats = (const float*)d_in[0];
    const float* proj_mat  = (const float*)d_in[1];
    const float* query     = (const float*)d_in[2];
    const float* query_pos = (const float*)d_in[3];
    const float* ref_pts   = (const float*)d_in[4];
    const float* mha_in_w  = (const float*)d_in[5];
    const float* mha_in_b  = (const float*)d_in[6];
    const float* mha_out_w = (const float*)d_in[7];
    const float* mha_out_b = (const float*)d_in[8];
    const float* off_w1    = (const float*)d_in[9];
    const float* off_b1    = (const float*)d_in[10];
    const float* off_w2    = (const float*)d_in[11];
    const float* off_b2    = (const float*)d_in[12];
    const float* am_w1     = (const float*)d_in[13];
    const float* am_b1     = (const float*)d_in[14];
    const float* am_w2     = (const float*)d_in[15];
    const float* am_b2     = (const float*)d_in[16];
    const float* t_ln1_g   = (const float*)d_in[17];
    const float* t_ln1_b   = (const float*)d_in[18];
    const float* t_in_w    = (const float*)d_in[19];
    const float* t_in_b    = (const float*)d_in[20];
    const float* t_out_w   = (const float*)d_in[21];
    const float* t_out_b   = (const float*)d_in[22];
    const float* t_ln2_g   = (const float*)d_in[23];
    const float* t_ln2_b   = (const float*)d_in[24];
    const float* t_ff_w1   = (const float*)d_in[25];
    const float* t_ff_b1   = (const float*)d_in[26];
    const float* t_ff_w2   = (const float*)d_in[27];
    const float* t_ff_b2   = (const float*)d_in[28];
    const float* ffn_w1    = (const float*)d_in[29];
    const float* ffn_b1    = (const float*)d_in[30];
    const float* ffn_w2    = (const float*)d_in[31];
    const float* ffn_b2    = (const float*)d_in[32];
    const float* op_w      = (const float*)d_in[33];
    const float* op_b      = (const float*)d_in[34];
    const float* pe_w1     = (const float*)d_in[35];
    const float* pe_b1     = (const float*)d_in[36];
    const float* pe_ln1_g  = (const float*)d_in[37];
    const float* pe_ln1_b  = (const float*)d_in[38];
    const float* pe_w2     = (const float*)d_in[39];
    const float* pe_b2     = (const float*)d_in[40];
    const float* pe_ln2_g  = (const float*)d_in[41];
    const float* pe_ln2_b  = (const float*)d_in[42];
    const float* ln1_g     = (const float*)d_in[43];
    const float* ln1_b     = (const float*)d_in[44];
    const float* ln2_g     = (const float*)d_in[45];
    const float* ln2_b     = (const float*)d_in[46];
    const float* ln3_g     = (const float*)d_in[47];
    const float* ln3_b     = (const float*)d_in[48];
    float* out = (float*)d_out;

    // workspace layout
    float* W = (float*)d_ws;
    size_t o = 0;
    auto alloc = [&](size_t n) { float* p = W + o; o += n; return p; };
    float* X    = alloc((size_t)TOK * E_DIM);        // sampled features [QP,CAMS,E]
    float* Y    = alloc((size_t)TOK * E_DIM);        // LN buffer
    float* QKV  = alloc((size_t)TOK * 3 * E_DIM);    // qkv / ffn hidden (phase1 qkv0 too)
    float* ATT  = alloc((size_t)TOK * E_DIM);        // attn out (phase1 probs too)
    float* q    = alloc((size_t)Q_NUM * E_DIM);
    float* qn   = alloc((size_t)Q_NUM * E_DIM);
    float* aout = alloc((size_t)Q_NUM * E_DIM);
    float* t0   = alloc((size_t)Q_NUM * E_DIM);
    float* t1   = alloc((size_t)Q_NUM * E_DIM);
    float* t2   = alloc((size_t)Q_NUM * E_DIM);
    float* offv = alloc((size_t)Q_NUM * 2 * P_PTS);
    float* amw  = alloc((size_t)Q_NUM * P_PTS);
    float* camxy= alloc((size_t)CAMS * QP * 2);
    unsigned char* validb = (unsigned char*)(W + o);

    const dim3 blk(256);
    auto ggrid = [](int M, int N) { return dim3((N + BN - 1) / BN, (M + BM - 1) / BM); };

    // ---- phase 1: self-attention over 900 queries ----
    k_add2<<<dim3((Q_NUM * E_DIM + 255) / 256), blk, 0, stream>>>(query, query_pos, q, Q_NUM * E_DIM);
    // qkv0 = q @ mha_in_w^T + b  -> QKV
    gemm_wmma<1,0,0><<<ggrid(Q_NUM, 3 * E_DIM), blk, 0, stream>>>(
        q, mha_in_w, mha_in_b, nullptr, QKV, Q_NUM, 3 * E_DIM, E_DIM, E_DIM, E_DIM, 3 * E_DIM, 1.0f);
    // per-head scores (into ATT) and softmax
    for (int h = 0; h < NH; ++h) {
        gemm_wmma<1,0,0><<<ggrid(Q_NUM, Q_NUM), blk, 0, stream>>>(
            QKV + h * HD, QKV + E_DIM + h * HD, nullptr, nullptr,
            ATT + (size_t)h * Q_NUM * Q_NUM,
            Q_NUM, Q_NUM, HD, 3 * E_DIM, 3 * E_DIM, Q_NUM, 0.125f);
    }
    k_softmax_rows<<<dim3(NH * Q_NUM), blk, 0, stream>>>(ATT, Q_NUM);
    // per-head  probs @ V  -> aout
    for (int h = 0; h < NH; ++h) {
        gemm_wmma<0,0,0><<<ggrid(Q_NUM, HD), blk, 0, stream>>>(
            ATT + (size_t)h * Q_NUM * Q_NUM, QKV + 2 * E_DIM + h * HD, nullptr, nullptr,
            aout + h * HD, Q_NUM, HD, Q_NUM, Q_NUM, 3 * E_DIM, E_DIM, 1.0f);
    }
    // out projection + residual, then LN -> qn
    gemm_wmma<1,0,1><<<ggrid(Q_NUM, E_DIM), blk, 0, stream>>>(
        aout, mha_out_w, mha_out_b, q, t0, Q_NUM, E_DIM, E_DIM, E_DIM, E_DIM, E_DIM, 1.0f);
    k_layernorm<<<dim3(Q_NUM), blk, 0, stream>>>(t0, nullptr, nullptr, ln1_g, ln1_b, qn, E_DIM, 0);

    // offset MLP: relu(qn@w1^T+b1) @ w2^T + b2
    gemm_wmma<1,1,0><<<ggrid(Q_NUM, E_DIM), blk, 0, stream>>>(
        qn, off_w1, off_b1, nullptr, Y, Q_NUM, E_DIM, E_DIM, E_DIM, E_DIM, E_DIM, 1.0f);
    k_small_fc<<<dim3((Q_NUM * 2 * P_PTS + 255) / 256), blk, 0, stream>>>(
        Y, off_w2, off_b2, offv, Q_NUM, 2 * P_PTS, E_DIM);
    // attention-weights MLP + softmax over P
    gemm_wmma<1,1,0><<<ggrid(Q_NUM, E_DIM), blk, 0, stream>>>(
        qn, am_w1, am_b1, nullptr, Y, Q_NUM, E_DIM, E_DIM, E_DIM, E_DIM, E_DIM, 1.0f);
    k_small_fc<<<dim3((Q_NUM * P_PTS + 255) / 256), blk, 0, stream>>>(
        Y, am_w2, am_b2, amw, Q_NUM, P_PTS, E_DIM);
    k_softmax_rows<<<dim3(Q_NUM), blk, 0, stream>>>(amw, P_PTS);

    // ---- feature sampling ----
    k_project<<<dim3((QP + 255) / 256), blk, 0, stream>>>(ref_pts, offv, proj_mat, camxy, validb);
    k_sample<<<dim3(CAMS * QP), dim3(64), 0, stream>>>(img_feats, camxy, X);

    // ---- ImgFeatureTransformer, depth 4 ----
    for (int i = 0; i < DEPTH; ++i) {
        const float* in_w  = t_in_w  + (size_t)i * 3 * E_DIM * E_DIM;
        const float* in_b  = t_in_b  + (size_t)i * 3 * E_DIM;
        const float* out_w = t_out_w + (size_t)i * E_DIM * E_DIM;
        const float* out_b = t_out_b + (size_t)i * E_DIM;
        const float* f1w   = t_ff_w1 + (size_t)i * E_DIM * E_DIM;
        const float* f1b   = t_ff_b1 + (size_t)i * E_DIM;
        const float* f2w   = t_ff_w2 + (size_t)i * E_DIM * E_DIM;
        const float* f2b   = t_ff_b2 + (size_t)i * E_DIM;

        k_layernorm<<<dim3(TOK), blk, 0, stream>>>(
            X, nullptr, nullptr, t_ln1_g + i * E_DIM, t_ln1_b + i * E_DIM, Y, E_DIM, 0);
        gemm_wmma<1,0,0><<<ggrid(TOK, 3 * E_DIM), blk, 0, stream>>>(
            Y, in_w, in_b, nullptr, QKV, TOK, 3 * E_DIM, E_DIM, E_DIM, E_DIM, 3 * E_DIM, 1.0f);
        k_tattn<<<dim3(QP * NH), dim3(64), 0, stream>>>(QKV, validb, ATT);
        gemm_wmma<1,0,1><<<ggrid(TOK, E_DIM), blk, 0, stream>>>(
            ATT, out_w, out_b, X, X, TOK, E_DIM, E_DIM, E_DIM, E_DIM, E_DIM, 1.0f);
        k_layernorm<<<dim3(TOK), blk, 0, stream>>>(
            X, nullptr, nullptr, t_ln2_g + i * E_DIM, t_ln2_b + i * E_DIM, Y, E_DIM, 0);
        gemm_wmma<1,1,0><<<ggrid(TOK, E_DIM), blk, 0, stream>>>(
            Y, f1w, f1b, nullptr, QKV, TOK, E_DIM, E_DIM, E_DIM, E_DIM, E_DIM, 1.0f);
        gemm_wmma<1,0,1><<<ggrid(TOK, E_DIM), blk, 0, stream>>>(
            QKV, f2w, f2b, X, X, TOK, E_DIM, E_DIM, E_DIM, E_DIM, E_DIM, 1.0f);
    }

    // ---- head ----
    k_reduce_out<<<dim3((Q_NUM * E_DIM + 255) / 256), blk, 0, stream>>>(X, amw, t0);
    gemm_wmma<1,0,0><<<ggrid(Q_NUM, E_DIM), blk, 0, stream>>>(
        t0, op_w, op_b, nullptr, t1, Q_NUM, E_DIM, E_DIM, E_DIM, E_DIM, E_DIM, 1.0f);

    // positional encoder branch
    k_pe1<<<dim3((Q_NUM * E_DIM + 255) / 256), blk, 0, stream>>>(ref_pts, pe_w1, pe_b1, t2);
    k_layernorm<<<dim3(Q_NUM), blk, 0, stream>>>(t2, nullptr, nullptr, pe_ln1_g, pe_ln1_b, t0, E_DIM, 1);
    gemm_wmma<1,0,0><<<ggrid(Q_NUM, E_DIM), blk, 0, stream>>>(
        t0, pe_w2, pe_b2, nullptr, t2, Q_NUM, E_DIM, E_DIM, E_DIM, E_DIM, E_DIM, 1.0f);
    k_layernorm<<<dim3(Q_NUM), blk, 0, stream>>>(t2, nullptr, nullptr, pe_ln2_g, pe_ln2_b, t0, E_DIM, 1);

    // qf = LN(out + qn + pe)
    k_layernorm<<<dim3(Q_NUM), blk, 0, stream>>>(t1, qn, t0, ln2_g, ln2_b, t2, E_DIM, 0);
    // final FFN + residual + LN -> d_out
    gemm_wmma<1,1,0><<<ggrid(Q_NUM, E_DIM), blk, 0, stream>>>(
        t2, ffn_w1, ffn_b1, nullptr, t0, Q_NUM, E_DIM, E_DIM, E_DIM, E_DIM, E_DIM, 1.0f);
    gemm_wmma<1,0,1><<<ggrid(Q_NUM, E_DIM), blk, 0, stream>>>(
        t0, ffn_w2, ffn_b2, t2, t1, Q_NUM, E_DIM, E_DIM, E_DIM, E_DIM, E_DIM, 1.0f);
    k_layernorm<<<dim3(Q_NUM), blk, 0, stream>>>(t1, nullptr, nullptr, ln3_g, ln3_b, out, E_DIM, 0);
}
